// AdditiveAttention_50070728737610
// MI455X (gfx1250) — matrix operand order repaired and checked by reading, not compiled, not run on hardware
//
#include <hip/hip_runtime.h>
#include <hip/hip_bf16.h>
#include <cstdint>

typedef __attribute__((ext_vector_type(16))) _Float16 v16h;
typedef __attribute__((ext_vector_type(8)))  float    v8f;

#define BZ   8
#define LSEQ 512
#define DDIM 512
#define HDIM 100
#define HPAD 104
#define TILE 16
#define NTJ  (LSEQ / TILE)   // 32 j-tiles

// ---------------------------------------------------------------------------
// Kernel 1: A = q @ W1^T ; B = q @ W2^T + b2   (per the reference, BOTH use q)
// One block = 16 rows of one batch. q rows staged in LDS; W rows from L2.
// ---------------------------------------------------------------------------
__global__ __launch_bounds__(256)
void proj_kernel(const float* __restrict__ q,
                 const float* __restrict__ W1,
                 const float* __restrict__ W2,
                 const float* __restrict__ b2,
                 float* __restrict__ Aout,
                 float* __restrict__ Bout) {
  __shared__ float qs[TILE][DDIM];                      // 32 KB
  const int bx  = blockIdx.x;
  const int b   = bx >> 5;                              // / (L/16)
  const int i0  = (bx & 31) * TILE;
  const int tid = threadIdx.x;

  const float4* qg  = (const float4*)(q + ((size_t)b * LSEQ + i0) * DDIM);
  float4*       qsv = (float4*)&qs[0][0];
  for (int t = tid; t < TILE * DDIM / 4; t += 256) qsv[t] = qg[t];
  __syncthreads();

  for (int t = tid; t < TILE * HDIM * 2; t += 256) {
    const int  sel = (t >= TILE * HDIM);                // 0 -> A, 1 -> B
    const int  idx = sel ? (t - TILE * HDIM) : t;
    const int  i   = idx / HDIM;
    const int  h   = idx % HDIM;
    const float4* wv = (const float4*)((sel ? W2 : W1) + (size_t)h * DDIM);
    const float4* qv = (const float4*)&qs[i][0];
    float acc = 0.f;
    for (int dd = 0; dd < DDIM / 4; ++dd) {
      const float4 a = qv[dd], w = wv[dd];
      acc += a.x * w.x + a.y * w.y + a.z * w.z + a.w * w.w;
    }
    const size_t o = ((size_t)b * LSEQ + i0 + i) * HDIM + h;
    if (sel) Bout[o] = acc + b2[h];
    else     Aout[o] = acc;
  }
}

// ---------------------------------------------------------------------------
// Kernel 2: fused score (tanh) + mask + softmax + (att @ v) via WMMA f16.
// One block = one (batch, 16-query tile). Never materializes the 4D tensor.
// ---------------------------------------------------------------------------
__global__ __launch_bounds__(256)
void attn_kernel(const float* __restrict__ Aproj,
                 const float* __restrict__ Bproj,
                 const float* __restrict__ w_out,
                 const uint8_t* __restrict__ mask,
                 const float* __restrict__ v,
                 float* __restrict__ out) {
  __shared__ float    As[TILE][HPAD];     // 6656 B
  __shared__ float    Bs[TILE][HPAD];     // 6656 B (reused as softmax scratch)
  __shared__ float    wo[HPAD];           //  416 B
  __shared__ float    sc[TILE][LSEQ];     // 32 KB  scores
  __shared__ _Float16 ph[TILE][LSEQ];     // 16 KB  softmax probs (f16)
  __shared__ float    rmax[TILE], rsum[TILE];

  const int bx    = blockIdx.x;
  const int b     = bx >> 5;
  const int i0    = (bx & 31) * TILE;
  const int tid   = threadIdx.x;
  const int lane  = tid & 31;
  const int wave  = tid >> 5;             // 8 waves of 32
  const int lhalf = lane >> 4;            // 0 | 1
  const int lm    = lane & 15;

  // ---- stage A tile + w_out ----
  for (int t = tid; t < TILE * HDIM; t += 256)
    As[t / HDIM][t % HDIM] =
        Aproj[((size_t)b * LSEQ + i0 + t / HDIM) * HDIM + (t % HDIM)];
  if (tid < HDIM) wo[tid] = w_out[tid];

  // ---- scores: 256 threads == the 16x16 (i,j) tile ----
  const int si = tid >> 4;                // query row in tile
  const int sj = tid & 15;                // key col in j-tile
  for (int jt = 0; jt < NTJ; ++jt) {
    __syncthreads();
    for (int t = tid; t < TILE * HDIM; t += 256)
      Bs[t / HDIM][t % HDIM] =
          Bproj[((size_t)b * LSEQ + jt * TILE + t / HDIM) * HDIM + (t % HDIM)];
    __syncthreads();
    float s = 0.f;
    for (int h = 0; h < HDIM; ++h)
      s += wo[h] * tanhf(As[si][h] + Bs[sj][h]);      // v_tanh_f32
    const int jg = jt * TILE + sj;
    if (!mask[(size_t)b * LSEQ + jg]) s = -1e9f;
    sc[si][jg] = s;
  }
  __syncthreads();

  // ---- softmax per row: 16 threads/row, Bs reused as reduction scratch ----
  {
    const int row = tid >> 4, c = tid & 15;
    float m = -3.0e38f;
    for (int j = c; j < LSEQ; j += 16) m = fmaxf(m, sc[row][j]);
    Bs[row][c] = m;
    __syncthreads();
    if (c == 0) {
      float mm = Bs[row][0];
      for (int x = 1; x < 16; ++x) mm = fmaxf(mm, Bs[row][x]);
      rmax[row] = mm;
    }
    __syncthreads();
    const float mm = rmax[row];
    float sum = 0.f;
    for (int j = c; j < LSEQ; j += 16) sum += __expf(sc[row][j] - mm);
    Bs[row][c] = sum;
    __syncthreads();
    if (c == 0) {
      float ss = 0.f;
      for (int x = 0; x < 16; ++x) ss += Bs[row][x];
      rsum[row] = 1.f / ss;
    }
    __syncthreads();
    const float inv = rsum[row];
    for (int j = c; j < LSEQ; j += 16)
      ph[row][j] = (_Float16)(__expf(sc[row][j] - mm) * inv);
  }
  __syncthreads();

  // ---- out[i, :] = probs @ v  via v_wmma_f32_16x16x32_f16 ----
  // Each wave owns 4 N-tiles of 16 columns; K marches over j in steps of 32.
  for (int nt = wave; nt < DDIM / TILE; nt += 8) {
    const int n0 = nt * TILE;
    v8f acc = {};
    for (int kt = 0; kt < LSEQ / 32; ++kt) {
      const int k0 = kt * 32;
      v16h afrag, bfrag;
      // A (16x32 f16) ISA layout: lane = M; elem e -> K = (e&7)+16*(e>>3)+8*lhalf
      #pragma unroll
      for (int e = 0; e < 16; ++e) {
        const int ka = (e & 7) + ((e >> 3) << 4) + (lhalf << 3);
        afrag[e] = ph[lm][k0 + ka];
      }
      // B (32x16 f16) ISA layout: lane = N (lm); elem e -> K = (e&7)+16*(e>>3)+8*lhalf
      const float* vcol =
          v + ((size_t)b * LSEQ + k0 + (lhalf << 3)) * DDIM + n0 + lm;
      __builtin_prefetch(vcol, 0, 1);                 // global_prefetch_b8
      #pragma unroll
      for (int e = 0; e < 16; ++e)
        bfrag[e] = (_Float16)vcol[(size_t)((e & 7) + ((e >> 3) << 4)) * DDIM];
      acc = __builtin_amdgcn_wmma_f32_16x16x32_f16(
          /*neg_a=*/false, afrag, /*neg_b=*/false, bfrag,
          /*c_mod=*/(short)0, acc, /*reuse_a=*/false, /*reuse_b=*/false);
    }
    // D layout: VGPR r -> row = r + 8*lhalf, col = n0 + lm
    #pragma unroll
    for (int r = 0; r < 8; ++r) {
      const int row = r + (lhalf << 3);
      out[((size_t)b * LSEQ + i0 + row) * DDIM + n0 + lm] = acc[r];
    }
  }
}

// ---------------------------------------------------------------------------
extern "C" void kernel_launch(void* const* d_in, const int* in_sizes, int n_in,
                              void* d_out, int out_size, void* d_ws, size_t ws_size,
                              hipStream_t stream) {
  const float*   q     = (const float*)d_in[0];
  // d_in[1] = k : intentionally unused (reference bug: k is never read)
  const float*   v     = (const float*)d_in[2];
  const uint8_t* mask  = (const uint8_t*)d_in[3];
  const float*   W1    = (const float*)d_in[4];
  const float*   W2    = (const float*)d_in[5];
  const float*   b2    = (const float*)d_in[6];
  const float*   w_out = (const float*)d_in[7];
  float*         out   = (float*)d_out;

  float* Aproj = (float*)d_ws;                         // 8*512*100 f32
  float* Bproj = Aproj + (size_t)BZ * LSEQ * HDIM;     // 8*512*100 f32

  dim3 grid(BZ * (LSEQ / TILE));                       // 256 blocks
  proj_kernel<<<grid, 256, 0, stream>>>(q, W1, W2, b2, Aproj, Bproj);
  attn_kernel<<<grid, 256, 0, stream>>>(Aproj, Bproj, w_out, mask, v, out);
}
